// DFTGN_AtomGraph_80547816669791
// MI455X (gfx1250) — matrix operand order repaired and checked by reading, not compiled, not run on hardware
//
#include <hip/hip_runtime.h>
#include <hip/hip_bf16.h>

// ---------------------------------------------------------------------------
// DFTGN AtomGraph forward for MI455X (gfx1250).
// Compute-bound on edge GEMMs (~220 GFLOP); all data L2-resident (192MB).
// bf16 WMMA (v_wmma_f32_16x16x32_bf16, f32 accumulate) for all GEMMs.
// This revision: 64-edge M-tiles with B-fragment reuse (4x less L2 weight
// traffic) and LDS tiles stored directly in WMMA A-fragment layout so each
// A fragment is a single 32B LDS read (2x ds_load_b128) instead of 8x b32.
// ---------------------------------------------------------------------------

typedef __attribute__((ext_vector_type(16))) __bf16 v16bf;
typedef __attribute__((ext_vector_type(8)))  float  v8f;

#define F   128
#define FE  384
#define K1  36
#define K1P 64
#define K2  26
#define K2P 32

#define MT_E 4     // 64 edges per workgroup
#define MT_N 2     // 32 nodes per workgroup
#define KC_FE (FE / 32)   // 12
#define KC_F  (F / 32)    // 4
#define KC_CS (K1P / 32)  // 2
#define KC_PW (K2P / 32)  // 1

static __device__ __forceinline__ float sigmoidf_(float x) {
    return 1.0f / (1.0f + __expf(-x));
}
static __device__ __forceinline__ float eluf_(float x) {
    return x > 0.0f ? x : (__expf(x) - 1.0f);
}
static __device__ __forceinline__ void atomic_add_f32(float* p, float v) {
    unsafeAtomicAdd(p, v);   // lowers to global_atomic_add_f32
}

static __device__ __forceinline__ v8f wmma_bf16(v16bf a, v16bf b, v8f c) {
    // D = A(16x32 bf16) * B(32x16 bf16) + C(16x16 f32)
    return __builtin_amdgcn_wmma_f32_16x16x32_bf16(
        false, a, false, b, (short)0, c, false, false);
}

// ---- A-fragment-layout LDS tiles --------------------------------------------
// ISA 16-bit A 16x32 layout: lane = half*16 + m; VGPR j holds
// K = (j>>2)*16 + half*8 + (j&3)*2 (+1). We store tiles pre-swizzled:
// lds[((mt*KC + kc)*32 + lane)*16 + j*2 + which].
static __device__ __forceinline__ void
store_a(__bf16* lds, int mt, int KC, int m_local, int k, float v) {
    int kc = k >> 5;
    int kk = k & 31;
    int h     = (kk >> 3) & 1;
    int j     = ((kk >> 4) << 2) | ((kk & 7) >> 1);
    int which = kk & 1;
    int lane  = (h << 4) | m_local;
    lds[((((mt * KC + kc) << 5) + lane) << 4) + (j << 1) + which] = (__bf16)v;
}

static __device__ __forceinline__ v16bf
load_a_frag(const __bf16* lds, int mt, int KC, int kc, int lane) {
    return *(const v16bf*)(lds + ((((mt * KC + kc) << 5) + lane) << 4));
}

// B fragment from pre-packed weights: 16 contiguous halfs per lane.
static __device__ __forceinline__ v16bf
load_b_frag(const __bf16* packed, int kc, int nt, int lane) {
    return *(const v16bf*)(packed + ((((kc << 3) + nt) << 5) + lane) * 16);
}

// Pack fp32 weight [K, 128] into B-fragment layout, bf16, K zero-padded to Kpad.
// ISA 16-bit B 32x16 layout: lane holds col n = lane&15;
// element s holds K = kc*32 + (s>>3)*16 + (lane>>4)*8 + (s&7).
__global__ void pack_w(const float* __restrict__ src, __bf16* __restrict__ dst,
                       int K, int Kpad) {
    int id = blockIdx.x * 256 + threadIdx.x;
    if (id >= Kpad * F) return;
    int j2   = id & 15;
    int lane = (id >> 4) & 31;
    int c    = id >> 9;              // kc*8 + nt
    int kc = c >> 3, nt = c & 7;
    int j = j2 >> 1, which = j2 & 1;
    int n = (nt << 4) + (lane & 15);
    int k = (kc << 5) + ((j2 >> 3) << 4) + ((lane >> 4) << 3) + (j2 & 7);
    float v = (k < K) ? src[k * F + n] : 0.0f;
    dst[id] = (__bf16)v;
}

// x = sigmoid(nodes @ W_emb) ; 32 nodes x 128 cols per block
__global__ void __launch_bounds__(128)
embed_kernel(const float* __restrict__ nodes, const __bf16* __restrict__ Wp,
             float* __restrict__ x_out) {
    __shared__ __attribute__((aligned(32))) __bf16 ldsX[MT_N * KC_F * 512];
    int n0 = blockIdx.x * (MT_N * 16);
    int tid = threadIdx.x;
#pragma unroll 4
    for (int i = 0; i < MT_N * 16; ++i)
        store_a(ldsX, i >> 4, KC_F, i & 15, tid, nodes[(size_t)(n0 + i) * F + tid]);
    __syncthreads();

    int wave = tid >> 5, lane = tid & 31;
    int mrow = lane & 15, half = lane >> 4;
#pragma unroll
    for (int t = 0; t < 2; ++t) {
        int nt = wave * 2 + t;
        v8f acc[MT_N] = {};
#pragma unroll
        for (int kc = 0; kc < KC_F; ++kc) {
            v16bf b = load_b_frag(Wp, kc, nt, lane);
#pragma unroll
            for (int mt = 0; mt < MT_N; ++mt)
                acc[mt] = wmma_bf16(load_a_frag(ldsX, mt, KC_F, kc, lane), b, acc[mt]);
        }
        int col = (nt << 4) + mrow;
#pragma unroll
        for (int mt = 0; mt < MT_N; ++mt)
#pragma unroll
            for (int r = 0; r < 8; ++r) {
                int m = (mt << 4) + r + (half << 3);
                x_out[(size_t)(n0 + m) * F + col] = sigmoidf_(acc[mt][r]);
            }
    }
}

// Fused edge block: gather -> fe -> {gate,mlp,z1,z2} GEMMs -> z -> atomic scatter
// 64 edges per workgroup; B fragments reused across 4 m-tiles.
__global__ void __launch_bounds__(128)
edge_kernel(const float* __restrict__ x_in,
            const int* __restrict__ esrc, const int* __restrict__ etgt,
            const float* __restrict__ edist,
            const float* __restrict__ cs, const float* __restrict__ pw,
            const __bf16* __restrict__ Wg, const __bf16* __restrict__ Wm,
            const __bf16* __restrict__ W1v, const __bf16* __restrict__ W2v,
            const __bf16* __restrict__ W2vg,
            float* __restrict__ x_out) {
    __shared__ __attribute__((aligned(32))) __bf16 ldsFE[MT_E * KC_FE * 512]; // 48KB
    __shared__ __attribute__((aligned(32))) __bf16 ldsCS[MT_E * KC_CS * 512]; //  8KB
    __shared__ __attribute__((aligned(32))) __bf16 ldsPW[MT_E * KC_PW * 512]; //  4KB
    __shared__ int lds_src[MT_E * 16];

    int e0 = blockIdx.x * (MT_E * 16);
    int tid = threadIdx.x;
    if (tid < MT_E * 16) lds_src[tid] = esrc[e0 + tid];

#pragma unroll 4
    for (int i = 0; i < MT_E * 16; ++i) {
        int   si  = esrc[e0 + i];                 // uniform -> scalar load
        int   tj  = etgt[e0 + i];
        float rin = 1.0f / edist[e0 + i];
        float ni  = x_in[(size_t)si * F + tid];
        float nj  = x_in[(size_t)tj * F + tid];
        int mt = i >> 4, ml = i & 15;
        store_a(ldsFE, mt, KC_FE, ml, tid,            ni);
        store_a(ldsFE, mt, KC_FE, ml, F + tid,        nj);
        store_a(ldsFE, mt, KC_FE, ml, 2 * F + tid,    (nj - ni) * rin);
    }
    for (int idx = tid; idx < MT_E * 16 * K1P; idx += 128) {
        int e = idx >> 6, k = idx & (K1P - 1);
        float v = (k < K1) ? cs[(size_t)(e0 + e) * K1 + k] : 0.0f;
        store_a(ldsCS, e >> 4, KC_CS, e & 15, k, v);
    }
    for (int idx = tid; idx < MT_E * 16 * K2P; idx += 128) {
        int e = idx >> 5, k = idx & (K2P - 1);
        float v = (k < K2) ? pw[(size_t)(e0 + e) * K2 + k] : 0.0f;
        store_a(ldsPW, e >> 4, KC_PW, e & 15, k, v);
    }
    __syncthreads();

    int wave = tid >> 5, lane = tid & 31;
    int mrow = lane & 15, half = lane >> 4;

#pragma unroll
    for (int t = 0; t < 2; ++t) {
        int nt = wave * 2 + t;
        v8f ag[MT_E] = {};
        v8f am[MT_E] = {};
#pragma unroll
        for (int kc = 0; kc < KC_FE; ++kc) {
            v16bf bg = load_b_frag(Wg, kc, nt, lane);
            v16bf bm = load_b_frag(Wm, kc, nt, lane);
#pragma unroll
            for (int mt = 0; mt < MT_E; ++mt) {
                v16bf af = load_a_frag(ldsFE, mt, KC_FE, kc, lane);
                ag[mt] = wmma_bf16(af, bg, ag[mt]);
                am[mt] = wmma_bf16(af, bm, am[mt]);
            }
        }
        int col = (nt << 4) + mrow;
#pragma unroll
        for (int mt = 0; mt < MT_E; ++mt) {
            v8f a1 = {}, a2 = {}, a2g = {};
            a1 = wmma_bf16(load_a_frag(ldsCS, mt, KC_CS, 0, lane),
                           load_b_frag(W1v, 0, nt, lane), a1);
            a1 = wmma_bf16(load_a_frag(ldsCS, mt, KC_CS, 1, lane),
                           load_b_frag(W1v, 1, nt, lane), a1);
            v16bf pf = load_a_frag(ldsPW, mt, KC_PW, 0, lane);
            a2  = wmma_bf16(pf, load_b_frag(W2v, 0, nt, lane), a2);
            a2g = wmma_bf16(pf, load_b_frag(W2vg, 0, nt, lane), a2g);
#pragma unroll
            for (int r = 0; r < 8; ++r) {
                int m = (mt << 4) + r + (half << 3);
                float z = sigmoidf_(ag[mt][r]) * eluf_(am[mt][r]) *
                          (a1[r] + a2[r] * sigmoidf_(a2g[r]));
                atomic_add_f32(&x_out[(size_t)lds_src[m] * F + col], z);
            }
        }
    }
}

// Fused node block: pool += elu(x@Wp1)*(x@Wp2) (per-graph atomic), x = elu(x@Wpsi)
// 32 nodes per workgroup; B fragments reused across 2 m-tiles.
__global__ void __launch_bounds__(128)
node_kernel(const float* __restrict__ x_in, const int* __restrict__ gidx,
            const __bf16* __restrict__ Wp1, const __bf16* __restrict__ Wp2,
            const __bf16* __restrict__ Wpsi,
            float* __restrict__ pool, float* __restrict__ x_out) {
    __shared__ __attribute__((aligned(32))) __bf16 ldsX[MT_N * KC_F * 512];
    __shared__ int lds_g[MT_N * 16];
    int n0 = blockIdx.x * (MT_N * 16);
    int tid = threadIdx.x;
    if (tid < MT_N * 16) lds_g[tid] = gidx[n0 + tid];
#pragma unroll 4
    for (int i = 0; i < MT_N * 16; ++i)
        store_a(ldsX, i >> 4, KC_F, i & 15, tid, x_in[(size_t)(n0 + i) * F + tid]);
    __syncthreads();

    int wave = tid >> 5, lane = tid & 31;
    int mrow = lane & 15, half = lane >> 4;
#pragma unroll
    for (int t = 0; t < 2; ++t) {
        int nt = wave * 2 + t;
        v8f p1[MT_N] = {};
        v8f p2[MT_N] = {};
        v8f ps[MT_N] = {};
#pragma unroll
        for (int kc = 0; kc < KC_F; ++kc) {
            v16bf b1 = load_b_frag(Wp1,  kc, nt, lane);
            v16bf b2 = load_b_frag(Wp2,  kc, nt, lane);
            v16bf bs = load_b_frag(Wpsi, kc, nt, lane);
#pragma unroll
            for (int mt = 0; mt < MT_N; ++mt) {
                v16bf af = load_a_frag(ldsX, mt, KC_F, kc, lane);
                p1[mt] = wmma_bf16(af, b1, p1[mt]);
                p2[mt] = wmma_bf16(af, b2, p2[mt]);
                ps[mt] = wmma_bf16(af, bs, ps[mt]);
            }
        }
        int col = (nt << 4) + mrow;
#pragma unroll
        for (int mt = 0; mt < MT_N; ++mt)
#pragma unroll
            for (int r = 0; r < 8; ++r) {
                int m = (mt << 4) + r + (half << 3);
                atomic_add_f32(&pool[(size_t)lds_g[m] * F + col],
                               eluf_(p1[mt][r]) * p2[mt][r]);
                x_out[(size_t)(n0 + m) * F + col] = eluf_(ps[mt][r]);
            }
    }
}

// out[g] = (elu(elu(pool @ Wlr1) @ Wlr2) @ Wlr3)  — tiny, VALU is fine
__global__ void __launch_bounds__(64)
readout_kernel(const float* __restrict__ pool,
               const float* __restrict__ Wlr1, const float* __restrict__ Wlr2,
               const float* __restrict__ Wlr3, float* __restrict__ out) {
    __shared__ float prow[128];
    __shared__ float h1[64];
    __shared__ float h2[42];
    int g = blockIdx.x, tid = threadIdx.x;
    prow[tid]      = pool[g * 128 + tid];
    prow[tid + 64] = pool[g * 128 + tid + 64];
    __syncthreads();
    {
        float s = 0.0f;
        for (int k = 0; k < 128; ++k) s += prow[k] * Wlr1[k * 64 + tid];
        h1[tid] = eluf_(s);
    }
    __syncthreads();
    if (tid < 42) {
        float s = 0.0f;
        for (int k = 0; k < 64; ++k) s += h1[k] * Wlr2[k * 42 + tid];
        h2[tid] = eluf_(s);
    }
    __syncthreads();
    if (tid == 0) {
        float s = 0.0f;
        for (int k = 0; k < 42; ++k) s += h2[k] * Wlr3[k];
        out[g] = s;
    }
}

extern "C" void kernel_launch(void* const* d_in, const int* in_sizes, int n_in,
                              void* d_out, int out_size, void* d_ws, size_t ws_size,
                              hipStream_t stream) {
    const float* nodes  = (const float*)d_in[0];
    const int*   esrc   = (const int*)d_in[1];
    const int*   etgt   = (const int*)d_in[2];
    const float* edist  = (const float*)d_in[3];
    const int*   gidx   = (const int*)d_in[4];
    /* d_in[5] node_counts: unused by reference forward */
    const float* cs     = (const float*)d_in[6];
    const float* pw     = (const float*)d_in[7];
    const float* W_emb  = (const float*)d_in[8];
    const float* Wg     = (const float*)d_in[9];
    const float* Wm     = (const float*)d_in[10];
    const float* W1v    = (const float*)d_in[11];
    const float* W2v    = (const float*)d_in[12];
    const float* W2vg   = (const float*)d_in[13];
    const float* Wp1    = (const float*)d_in[14];
    const float* Wp2    = (const float*)d_in[15];
    const float* Wpsi   = (const float*)d_in[16];
    const float* Wlr1   = (const float*)d_in[17];
    const float* Wlr2   = (const float*)d_in[18];
    const float* Wlr3   = (const float*)d_in[19];

    const int Nn = in_sizes[0] / F;   // 20000
    const int E  = in_sizes[1];       // 320000
    const int G  = in_sizes[5];       // 200

    char* ws = (char*)d_ws;
    size_t off = 0;
    auto wsalloc = [&](size_t bytes) -> char* {
        char* p = ws + off;
        off += (bytes + 255) & ~(size_t)255;
        return p;
    };
    float*  x_cur  = (float*)wsalloc((size_t)Nn * F * 4);
    float*  x_next = (float*)wsalloc((size_t)Nn * F * 4);
    float*  pool   = (float*)wsalloc((size_t)G * F * 4);
    __bf16* Wemb_p = (__bf16*)wsalloc((size_t)F * F * 2);
    __bf16* Wg_p   = (__bf16*)wsalloc((size_t)3 * FE * F * 2);
    __bf16* Wm_p   = (__bf16*)wsalloc((size_t)3 * FE * F * 2);
    __bf16* W1v_p  = (__bf16*)wsalloc((size_t)3 * K1P * F * 2);
    __bf16* W2v_p  = (__bf16*)wsalloc((size_t)3 * K2P * F * 2);
    __bf16* W2vg_p = (__bf16*)wsalloc((size_t)3 * K2P * F * 2);
    __bf16* Wp1_p  = (__bf16*)wsalloc((size_t)3 * F * F * 2);
    __bf16* Wp2_p  = (__bf16*)wsalloc((size_t)3 * F * F * 2);
    __bf16* Wpsi_p = (__bf16*)wsalloc((size_t)3 * F * F * 2);

    auto pack = [&](const float* src, __bf16* dst, int K, int Kpad) {
        int total = Kpad * F;
        pack_w<<<(total + 255) / 256, 256, 0, stream>>>(src, dst, K, Kpad);
    };
    pack(W_emb, Wemb_p, F, F);
    for (int i = 0; i < 3; ++i) {
        pack(Wg   + (size_t)i * FE * F, Wg_p   + (size_t)i * FE * F,  FE, FE);
        pack(Wm   + (size_t)i * FE * F, Wm_p   + (size_t)i * FE * F,  FE, FE);
        pack(W1v  + (size_t)i * K1 * F, W1v_p  + (size_t)i * K1P * F, K1, K1P);
        pack(W2v  + (size_t)i * K2 * F, W2v_p  + (size_t)i * K2P * F, K2, K2P);
        pack(W2vg + (size_t)i * K2 * F, W2vg_p + (size_t)i * K2P * F, K2, K2P);
        pack(Wp1  + (size_t)i * F * F,  Wp1_p  + (size_t)i * F * F,   F, F);
        pack(Wp2  + (size_t)i * F * F,  Wp2_p  + (size_t)i * F * F,   F, F);
        pack(Wpsi + (size_t)i * F * F,  Wpsi_p + (size_t)i * F * F,   F, F);
    }

    hipMemsetAsync(pool, 0, (size_t)G * F * 4, stream);

    embed_kernel<<<Nn / (MT_N * 16), 128, 0, stream>>>(nodes, Wemb_p, x_cur);

    for (int i = 0; i < 3; ++i) {
        // x_next = x_cur ; then scatter-add z into x_next (reference clone+index_add_)
        hipMemcpyAsync(x_next, x_cur, (size_t)Nn * F * 4, hipMemcpyDeviceToDevice, stream);
        edge_kernel<<<E / (MT_E * 16), 128, 0, stream>>>(
            x_cur, esrc, etgt, edist, cs, pw,
            Wg_p + (size_t)i * FE * F, Wm_p + (size_t)i * FE * F,
            W1v_p + (size_t)i * K1P * F, W2v_p + (size_t)i * K2P * F,
            W2vg_p + (size_t)i * K2P * F, x_next);
        node_kernel<<<Nn / (MT_N * 16), 128, 0, stream>>>(
            x_next, gidx,
            Wp1_p + (size_t)i * F * F, Wp2_p + (size_t)i * F * F,
            Wpsi_p + (size_t)i * F * F, pool, x_cur);
    }

    readout_kernel<<<G, 64, 0, stream>>>(pool, Wlr1, Wlr2, Wlr3, (float*)d_out);
}
